// PocketP450_18193481466586
// MI455X (gfx1250) — matrix operand order repaired and checked by reading, not compiled, not run on hardware
//
#include <hip/hip_runtime.h>
#include <hip/hip_bf16.h>

#define N_LIG   10000
#define N_PROT  40000
#define N_NODES 50000
#define N_EDGES 400000
#define NB      512
#define F_LIG   74
#define F_PROT  1280
#define HEADS   4
#define HDIM    64
#define HD      256

typedef __attribute__((ext_vector_type(16))) __bf16 v16bf;
typedef __attribute__((ext_vector_type(8)))  float  v8f;

// ---------------------------------------------------------------------------
// WMMA GEMM: C[M,HD] = A[M,K] @ W[K,HD], bf16 operands, f32 accumulate.
// K is a compile-time template parameter (74/256/1280). M must be a multiple
// of 16 (true for 10000/40000/50000) -> no row guards, straight-line stores.
// Tiles staged through LDS (coalesced global_load_b128 -> bf16 -> ds_store,
// ds_load_b128 in WMMA fragment layout). Block = 128 threads = 4 waves; each
// wave owns one 64-column strip == one attention head, so if asrc/adst are
// given, att_s/att_d are computed from the accumulators with a 16-lane
// shfl_xor reduction (no re-read of h from HBM).
// ---------------------------------------------------------------------------
template <int K>
__global__ __launch_bounds__(128)
void gemm_bf16_wmma(const float* __restrict__ A, const float* __restrict__ Bw,
                    float* __restrict__ C, int M,
                    const float* __restrict__ asrc, const float* __restrict__ adst,
                    float* __restrict__ att_s, float* __restrict__ att_d) {
  __shared__ __bf16 As[16][40];    // 16x32 A tile, rows padded to 80 B
  __shared__ __bf16 Bs[256][40];   // B tile transposed: [col][k], padded

  const int t    = threadIdx.x;
  const int lane = t & 31;
  const int wave = t >> 5;          // 0..3  == head id
  const int m0   = blockIdx.x * 16;
  const int n0   = wave * 64;
  const int half = lane >> 4;       // 0/1
  const int l16  = lane & 15;

  v8f acc[4] = {};

  for (int kb = 0; kb < K; kb += 32) {
    // ---- stage A tile (16 rows x 32 K): 4 f32 per thread ----
    {
      int r  = t >> 3;              // 0..15
      int c4 = (t & 7) * 4;         // 0..28
      const float* ap = A + (size_t)(m0 + r) * K;
#pragma unroll
      for (int j = 0; j < 4; ++j) {
        int k  = kb + c4 + j;
        int kk = (k < K) ? k : (K - 1);                // clamped address
        float v = ap[kk];
        As[r][c4 + j] = (__bf16)((k < K) ? v : 0.0f);  // value select only
      }
    }
    // ---- stage B tile (32 K x 256 cols), transposed into Bs[col][k] ----
#pragma unroll
    for (int i = 0; i < 16; ++i) {
      int flat4 = t + i * 128;       // 0..2047 float4s
      int idx   = flat4 * 4;
      int kr    = idx >> 8;          // 0..31
      int col   = idx & 255;         // multiple of 4
      int k     = kb + kr;
      int kk    = (k < K) ? k : (K - 1);
      const float* bp = Bw + (size_t)kk * HD + col;
      if (i == 0) {                  // branchless prefetch of next K tile
        int kn  = k + 32;
        int knn = (kn < K) ? kn : (K - 1);
        __builtin_prefetch(Bw + (size_t)knn * HD + col, 0, 1);
      }
      float4 v4 = *reinterpret_cast<const float4*>(bp);
      if (k >= K) { v4.x = 0.f; v4.y = 0.f; v4.z = 0.f; v4.w = 0.f; }
      Bs[col + 0][kr] = (__bf16)v4.x;
      Bs[col + 1][kr] = (__bf16)v4.y;
      Bs[col + 2][kr] = (__bf16)v4.z;
      Bs[col + 3][kr] = (__bf16)v4.w;
    }
    __syncthreads();

    // ---- A fragment: ISA 7.12.2 16-bit A 16x32 layout ----
    v16bf a;
#pragma unroll
    for (int e = 0; e < 8; ++e) {
      a[e]     = As[l16][half * 8 + e];        // K 0..7  / 8..15
      a[e + 8] = As[l16][16 + half * 8 + e];   // K 16..23 / 24..31
    }

    // ---- 4 B fragments (32x16 each) + WMMA ----
#pragma unroll
    for (int tt = 0; tt < 4; ++tt) {
      int col = n0 + tt * 16 + l16;
      v16bf b;
#pragma unroll
      for (int e = 0; e < 8; ++e) {
        b[e]     = Bs[col][half * 8 + e];        // K 0..7  / 8..15
        b[e + 8] = Bs[col][16 + half * 8 + e];   // K 16..23 / 24..31
      }
      acc[tt] = __builtin_amdgcn_wmma_f32_16x16x32_bf16(
          false, a, false, b, (short)0, acc[tt], false, false);
    }
    __syncthreads();
  }

  // ---- C/D: lanes 0-15 -> rows m0+0..7, lanes 16-31 -> rows m0+8..15 ----
#pragma unroll
  for (int tt = 0; tt < 4; ++tt)
#pragma unroll
    for (int r = 0; r < 8; ++r)
      C[(size_t)(m0 + half * 8 + r) * HD + n0 + tt * 16 + l16] = acc[tt][r];

  // ---- fused attention projections: head == wave, dims == this strip ----
  if (asrc != nullptr) {
    float as_v[4], ad_v[4];
#pragma unroll
    for (int tt = 0; tt < 4; ++tt) {
      as_v[tt] = asrc[n0 + tt * 16 + l16];
      ad_v[tt] = adst[n0 + tt * 16 + l16];
    }
#pragma unroll
    for (int r = 0; r < 8; ++r) {
      float s = 0.f, d2 = 0.f;
#pragma unroll
      for (int tt = 0; tt < 4; ++tt) {
        s  += acc[tt][r] * as_v[tt];
        d2 += acc[tt][r] * ad_v[tt];
      }
      // reduce across the 16 lanes of this half-group (masks < 16 keep the
      // two row-groups independent)
#pragma unroll
      for (int m = 1; m < 16; m <<= 1) {
        s  += __shfl_xor(s,  m, 32);
        d2 += __shfl_xor(d2, m, 32);
      }
      if (l16 == 0) {
        int row = m0 + half * 8 + r;
        att_s[row * HEADS + wave] = s;
        att_d[row * HEADS + wave] = d2;
      }
    }
  }
}

// ---------------------------------------------------------------------------
// Scatter / softmax phases: bulk VALU + device-scope atomics (L2-resident:
// whole working set ~160 MB < 192 MB L2).
// ---------------------------------------------------------------------------
__global__ void fill_kernel(float* __restrict__ p, float v, long n) {
  long i = (long)blockIdx.x * blockDim.x + threadIdx.x;
  if (i < n) p[i] = v;
}

__device__ inline void atomicMaxFloat(float* addr, float val) {
  int* ia = (int*)addr;
  int cur = __float_as_int(__builtin_nontemporal_load(addr));
  while (__int_as_float(cur) < val) {
    int assumed = cur;
    cur = atomicCAS(ia, assumed, __float_as_int(val));
    if (cur == assumed) break;
  }
}

__global__ void edge_max_kernel(const float* __restrict__ att_s,
                                const float* __restrict__ att_d,
                                const int* __restrict__ esrc,
                                const int* __restrict__ edst,
                                float* __restrict__ mbuf) {
  int idx = blockIdx.x * blockDim.x + threadIdx.x;
  if (idx >= N_EDGES * HEADS) return;
  int e = idx >> 2, hh = idx & 3;
  float v = att_s[esrc[e] * HEADS + hh] + att_d[edst[e] * HEADS + hh];
  v = v > 0.f ? v : 0.2f * v;                       // leaky_relu(0.2)
  atomicMaxFloat(&mbuf[edst[e] * HEADS + hh], v);
}

__global__ void edge_w_kernel(const float* __restrict__ att_s,
                              const float* __restrict__ att_d,
                              const int* __restrict__ esrc,
                              const int* __restrict__ edst,
                              const float* __restrict__ mbuf,
                              float* __restrict__ denom,
                              float* __restrict__ wbuf) {
  int idx = blockIdx.x * blockDim.x + threadIdx.x;
  if (idx >= N_EDGES * HEADS) return;
  int e = idx >> 2, hh = idx & 3;
  int d = edst[e];
  float v = att_s[esrc[e] * HEADS + hh] + att_d[d * HEADS + hh];
  v = v > 0.f ? v : 0.2f * v;
  float w = __expf(v - mbuf[d * HEADS + hh]);
  wbuf[idx] = w;
  atomicAdd(&denom[d * HEADS + hh], w);
}

__global__ void edge_agg_kernel(const float* __restrict__ h,
                                const float* __restrict__ wbuf,
                                const int* __restrict__ esrc,
                                const int* __restrict__ edst,
                                float* __restrict__ agg) {
  int e = blockIdx.x;        // one block per edge
  int c = threadIdx.x;       // channel 0..255
  int s = esrc[e], d = edst[e];
  float w = wbuf[e * HEADS + (c >> 6)];
  atomicAdd(&agg[(size_t)d * HD + c], h[(size_t)s * HD + c] * w);
}

__global__ void finalize_kernel(float* __restrict__ agg,
                                const float* __restrict__ denom) {
  int n = blockIdx.x, c = threadIdx.x;
  float v = agg[(size_t)n * HD + c] / (denom[n * HEADS + (c >> 6)] + 1e-9f);
  agg[(size_t)n * HD + c] = v > 0.f ? v : (__expf(v) - 1.0f);   // ELU
}

__global__ void pool_kernel(const float* __restrict__ h,
                            const int* __restrict__ gid,
                            float* __restrict__ pooled,
                            float* __restrict__ cnt) {
  int n = blockIdx.x, c = threadIdx.x;
  int g = gid[n];
  atomicAdd(&pooled[(size_t)g * HD + c], h[(size_t)n * HD + c]);
  if (c == 0) atomicAdd(&cnt[g], 1.0f);
}

__global__ void out_kernel(const float* __restrict__ pooled,
                           const float* __restrict__ cnt,
                           const float* __restrict__ scores,
                           const float* __restrict__ W_out,
                           const float* __restrict__ b_out,
                           float* __restrict__ out) {
  int b = blockIdx.x * blockDim.x + threadIdx.x;
  if (b >= NB) return;
  float c = cnt[b]; c = c > 1.0f ? c : 1.0f;
  float s = 0.f;
#pragma unroll 8
  for (int i = 0; i < HD; ++i) s += (pooled[(size_t)b * HD + i] / c) * W_out[i];
  s += scores[b] * W_out[HD] + b_out[0];
  out[b] = s;
}

// ---------------------------------------------------------------------------
static void run_gat_layer(const float* xin, float* h, float* agg,
                          const float* W, const float* asrc, const float* adst,
                          const int* esrc, const int* edst,
                          float* att_s, float* att_d, float* mbuf,
                          float* denom, float* wbuf, hipStream_t stream) {
  // GEMM with fused attention-dot epilogue (att_s/att_d from accumulators).
  gemm_bf16_wmma<HD><<<(N_NODES + 15) / 16, 128, 0, stream>>>(
      xin, W, h, N_NODES, asrc, adst, att_s, att_d);
  fill_kernel<<<(N_NODES * HEADS + 255) / 256, 256, 0, stream>>>(mbuf, -3.0e38f, (long)N_NODES * HEADS);
  fill_kernel<<<(N_NODES * HEADS + 255) / 256, 256, 0, stream>>>(denom, 0.f, (long)N_NODES * HEADS);
  fill_kernel<<<((long)N_NODES * HD + 255) / 256, 256, 0, stream>>>(agg, 0.f, (long)N_NODES * HD);
  edge_max_kernel<<<(N_EDGES * HEADS + 255) / 256, 256, 0, stream>>>(att_s, att_d, esrc, edst, mbuf);
  edge_w_kernel<<<(N_EDGES * HEADS + 255) / 256, 256, 0, stream>>>(att_s, att_d, esrc, edst, mbuf, denom, wbuf);
  edge_agg_kernel<<<N_EDGES, HD, 0, stream>>>(h, wbuf, esrc, edst, agg);
  finalize_kernel<<<N_NODES, HD, 0, stream>>>(agg, denom);
}

extern "C" void kernel_launch(void* const* d_in, const int* in_sizes, int n_in,
                              void* d_out, int out_size, void* d_ws, size_t ws_size,
                              hipStream_t stream) {
  const float* ligand_x  = (const float*)d_in[0];
  const float* protein_x = (const float*)d_in[1];
  const float* W_lig     = (const float*)d_in[2];
  const float* W_prot    = (const float*)d_in[3];
  const float* W1        = (const float*)d_in[4];
  const float* a1_src    = (const float*)d_in[5];
  const float* a1_dst    = (const float*)d_in[6];
  const float* W2        = (const float*)d_in[7];
  const float* a2_src    = (const float*)d_in[8];
  const float* a2_dst    = (const float*)d_in[9];
  const float* W_out     = (const float*)d_in[10];
  const float* b_out     = (const float*)d_in[11];
  const int*   esrc      = (const int*)d_in[12];
  const int*   edst      = (const int*)d_in[13];
  const int*   gid       = (const int*)d_in[14];
  const float* scores    = (const float*)d_in[15];
  float* out = (float*)d_out;

  // Workspace layout (floats). 3 node-feature buffers ping-pong across layers.
  float* ws     = (float*)d_ws;
  float* xbuf   = ws;                                    // [N,256]
  float* hbuf   = xbuf   + (size_t)N_NODES * HD;         // [N,256]
  float* aggbuf = hbuf   + (size_t)N_NODES * HD;         // [N,256]
  float* att_s  = aggbuf + (size_t)N_NODES * HD;         // [N,4]
  float* att_d  = att_s  + (size_t)N_NODES * HEADS;      // [N,4]
  float* mbuf   = att_d  + (size_t)N_NODES * HEADS;      // [N,4]
  float* denom  = mbuf   + (size_t)N_NODES * HEADS;      // [N,4]
  float* wbuf   = denom  + (size_t)N_NODES * HEADS;      // [E,4]
  float* pooled = wbuf   + (size_t)N_EDGES * HEADS;      // [B,256]
  float* cnt    = pooled + (size_t)NB * HD;              // [B]

  // 1) Input projections (bandwidth-bound: protein_x is 205 MB -> ~9us floor).
  gemm_bf16_wmma<F_LIG><<<(N_LIG + 15) / 16, 128, 0, stream>>>(
      ligand_x, W_lig, xbuf, N_LIG, nullptr, nullptr, nullptr, nullptr);
  gemm_bf16_wmma<F_PROT><<<(N_PROT + 15) / 16, 128, 0, stream>>>(
      protein_x, W_prot, xbuf + (size_t)N_LIG * HD, N_PROT,
      nullptr, nullptr, nullptr, nullptr);

  // 2) GAT layer 1: x(xbuf) -> h(hbuf) -> out(aggbuf, in place after finalize)
  run_gat_layer(xbuf, hbuf, aggbuf, W1, a1_src, a1_dst, esrc, edst,
                att_s, att_d, mbuf, denom, wbuf, stream);

  // 3) GAT layer 2: x(aggbuf) -> h(xbuf) -> out(hbuf)
  run_gat_layer(aggbuf, xbuf, hbuf, W2, a2_src, a2_dst, esrc, edst,
                att_s, att_d, mbuf, denom, wbuf, stream);

  // 4) Mean pool per graph + final linear.
  fill_kernel<<<((long)NB * HD + 255) / 256, 256, 0, stream>>>(pooled, 0.f, (long)NB * HD);
  fill_kernel<<<(NB + 255) / 256, 256, 0, stream>>>(cnt, 0.f, (long)NB);
  pool_kernel<<<N_NODES, HD, 0, stream>>>(hbuf, gid, pooled, cnt);
  out_kernel<<<(NB + 255) / 256, 256, 0, stream>>>(pooled, cnt, scores, W_out, b_out, out);
}
